// SSD_1821066133999
// MI455X (gfx1250) — matrix operand order repaired and checked by reading, not compiled, not run on hardware
//
#include <hip/hip_runtime.h>
#include <hip/hip_bf16.h>

#define D_MODEL 1024
#define D_STATE 64
#define D_INNER 1024
#define SEQ     4096
#define BATCH   4
#define NROW    (BATCH * SEQ)                 // 16384
#define XBC_W   (2 * D_INNER + 2 * D_STATE)   // 2176
#define CHUNK   64
#define NCHUNK  (SEQ / CHUNK)                 // 64
#define CG      32                            // channels per scan workgroup
#define NCG     (D_INNER / CG)                // 32

typedef __attribute__((ext_vector_type(16))) __bf16       v16bf;
typedef __attribute__((ext_vector_type(8)))  float        v8f;
typedef __attribute__((ext_vector_type(4)))  unsigned int u32x4;
typedef __attribute__((ext_vector_type(2)))  unsigned int u32x2;
typedef __attribute__((ext_vector_type(4)))  float        f32x4;
typedef __attribute__((ext_vector_type(4)))  int          i32x4;

#define AS1 __attribute__((address_space(1)))
#define AS3 __attribute__((address_space(3)))

#if defined(__has_builtin)
#if __has_builtin(__builtin_amdgcn_global_load_async_to_lds_b128)
#define HAVE_ASYNC_LDS 1
#endif
#endif
#ifndef HAVE_ASYNC_LDS
#define HAVE_ASYNC_LDS 0
#endif

#if HAVE_ASYNC_LDS
// signature (probe-confirmed by clang diagnostic): (int4 as1*, int4 as3*, imm off, imm cpol)
#define ASYNC_CP16(gp, lp) \
  __builtin_amdgcn_global_load_async_to_lds_b128((AS1 i32x4*)(gp), (AS3 i32x4*)(lp), 0, 0)
#endif

__device__ inline void async_wait0() {
#if defined(__has_builtin) && __has_builtin(__builtin_amdgcn_s_wait_asynccnt)
  __builtin_amdgcn_s_wait_asynccnt(0);
#else
  asm volatile("s_wait_asynccnt 0x0" ::: "memory");
#endif
}

union V16 {
  v16bf v;
  u32x4 q[2];
};

__device__ inline unsigned short f2bf(float f) {
  unsigned int u = __float_as_uint(f);
  u += 0x7FFFu + ((u >> 16) & 1u);              // round-to-nearest-even
  return (unsigned short)(u >> 16);
}
__device__ inline float bf2f(unsigned short h) {
  return __uint_as_float(((unsigned int)h) << 16);
}

// ---------------------------------------------------------------------------
// fp32 -> bf16 pack (4 elems/thread), n multiple of 4
// ---------------------------------------------------------------------------
__global__ __launch_bounds__(256)
void cvt_bf16(const float* __restrict__ src, unsigned short* __restrict__ dst, long n) {
  long i = ((long)blockIdx.x * 256 + threadIdx.x) * 4;
  if (i >= n) return;
  f32x4 v = *(const f32x4*)&src[i];
  u32x2 pk;
  pk.x = (unsigned int)f2bf(v.x) | ((unsigned int)f2bf(v.y) << 16);
  pk.y = (unsigned int)f2bf(v.z) | ((unsigned int)f2bf(v.w) << 16);
  *(u32x2*)&dst[i] = pk;
}

// ---------------------------------------------------------------------------
// C[M,N] = A[M,K] * B[N,K]^T   (bf16 in, fp32 out)
// block 256 (8 waves), block tile 128x128, k-tile 32; wave = 64x32.
// Double-buffered LDS staged via GLOBAL_LOAD_ASYNC_TO_LDS_B128:
// issue next k-tile's async copies, compute current tile, then
// s_wait_asynccnt + barrier — WMMA hides the async latency.
// ---------------------------------------------------------------------------
__global__ __launch_bounds__(256)
void wmma_gemm_bf16(const unsigned short* __restrict__ A, int lda,
                    const unsigned short* __restrict__ B, int ldb,
                    float* __restrict__ C, int ldc, int K) {
  __shared__ unsigned short As[2][128][48];   // 96B row stride, 16B-aligned chunks
  __shared__ unsigned short Bs[2][128][48];

  const int t    = threadIdx.x;
  const int lane = t & 31;
  const int wave = t >> 5;
  const int wm   = wave >> 2;   // 0..1
  const int wn   = wave & 3;    // 0..3
  const int hf   = lane >> 4;
  const int ql   = lane & 15;
  const long m0  = (long)blockIdx.y * 128;
  const long n0  = (long)blockIdx.x * 128;

  auto stage = [&](int k0, int buf) {
#pragma unroll
    for (int p = 0; p < 2; ++p) {
      int chunk = p * 256 + t;            // 0..511 16B-chunks per matrix
      int r  = chunk >> 2;
      int c8 = (chunk & 3) * 8;
#if HAVE_ASYNC_LDS
      ASYNC_CP16(&A[(m0 + r) * lda + k0 + c8], &As[buf][r][c8]);
      ASYNC_CP16(&B[(n0 + r) * ldb + k0 + c8], &Bs[buf][r][c8]);
#else
      *(u32x4*)&As[buf][r][c8] = *(const u32x4*)&A[(m0 + r) * lda + k0 + c8];
      *(u32x4*)&Bs[buf][r][c8] = *(const u32x4*)&B[(n0 + r) * ldb + k0 + c8];
#endif
    }
  };

  v8f acc[4][2] = {};

  stage(0, 0);
#if HAVE_ASYNC_LDS
  async_wait0();
#endif
  __syncthreads();

  for (int k0 = 0; k0 < K; k0 += 32) {
    const int cur = (k0 >> 5) & 1;
    if (k0 + 32 < K) stage(k0 + 32, cur ^ 1);   // prefetch next tile (async)

    V16 bfr[2];
#pragma unroll
    for (int j = 0; j < 2; ++j) {
      int nrow = wn * 32 + j * 16 + ql;
      const u32x4* bp = (const u32x4*)&Bs[cur][nrow][hf * 16];
      bfr[j].q[0] = bp[0];
      bfr[j].q[1] = bp[1];
    }
#pragma unroll
    for (int i = 0; i < 4; ++i) {
      int mrow = wm * 64 + i * 16 + ql;
      V16 afr;
      afr.q[0] = *(const u32x4*)&As[cur][mrow][hf * 16];
      afr.q[1] = *(const u32x4*)&As[cur][mrow][hf * 16 + 8];
#pragma unroll
      for (int j = 0; j < 2; ++j) {
        acc[i][j] = __builtin_amdgcn_wmma_f32_16x16x32_bf16(
            false, afr.v, false, bfr[j].v, (short)0, acc[i][j], false, false);
      }
    }
#if HAVE_ASYNC_LDS
    async_wait0();
#endif
    __syncthreads();
  }

#pragma unroll
  for (int i = 0; i < 4; ++i)
#pragma unroll
    for (int j = 0; j < 2; ++j)
#pragma unroll
      for (int r = 0; r < 8; ++r) {
        long m = m0 + wm * 64 + i * 16 + r + 8 * hf;
        long n = n0 + wn * 32 + j * 16 + ql;
        C[m * ldc + n] = acc[i][j][r];
      }
}

// ---------------------------------------------------------------------------
// depthwise causal conv (K=4) + bias + SiLU over the xc slice of xBC
// ---------------------------------------------------------------------------
__global__ __launch_bounds__(256)
void conv_silu(const float* __restrict__ xBC, const float* __restrict__ w,
               const float* __restrict__ bias, float* __restrict__ out) {
  long idx = (long)blockIdx.x * 256 + threadIdx.x;   // over NROW * D_INNER
  int  c   = (int)(idx & (D_INNER - 1));
  long row = idx >> 10;
  int  l   = (int)(row & (SEQ - 1));
  f32x4 wc = *(const f32x4*)&w[c * 4];
  float wv[4] = {wc.x, wc.y, wc.z, wc.w};
  float s = bias[c];
#pragma unroll
  for (int tp = 0; tp < 4; ++tp) {
    int ls = l - 3 + tp;
    if (ls >= 0) s += wv[tp] * xBC[(row - 3 + tp) * XBC_W + c];
  }
  out[idx] = s / (1.f + __expf(-s));
}

// ---------------------------------------------------------------------------
// Chunked SSD scan (Q=64). One block per (batch, 32-channel group) = 128 WGs.
//   G   = tril( (C .* a^{i+1}) @ (B .* a^{-(j+1)})^T )
//   Y   = [C~ | G] @ [H ; X]                   (WMMA, fused inter+intra)
//   H'  = a^Q .* H + (B .* a^{Q-1-j})^T @ X    (WMMA)
// ---------------------------------------------------------------------------
__global__ __launch_bounds__(256)
void ssd_scan(const float* __restrict__ xBC, const float* __restrict__ xconv,
              const float* __restrict__ A_param, float* __restrict__ yssd) {
  __shared__ unsigned short Xh[CG][128];      // [c][k]: k<64 -> H(n), k>=64 -> X(j)
  __shared__ unsigned short Ach[CHUNK][128];  // [i][k]: k<64 -> C~,   k>=64 -> G
  __shared__ unsigned short Btmp[CHUNK][64];  // B-hat[j][n], then Bbar^T[n][j]
  __shared__ unsigned short dPos[CHUNK][64];  // a_n^{i+1}
  __shared__ unsigned short dNeg[CHUNK][64];  // a_n^{-(j+1)}  (<= e^64, bf16 ok)
  __shared__ float          dQ[64];           // a_n^{64}

  const int t    = threadIdx.x;
  const int lane = t & 31;
  const int wave = t >> 5;
  const int hf   = lane >> 4;
  const int ql   = lane & 15;
  const int b    = blockIdx.x / NCG;
  const int cg0  = (blockIdx.x % NCG) * CG;
  const long rowBase = (long)b * SEQ;

  for (int e = t; e < CHUNK * 64; e += 256) {
    int i = e >> 6, n = e & 63;
    float en = __expf(A_param[n]);            // A_disc = exp(-exp(A_param[n]))
    dPos[i][n] = f2bf(__expf(-(float)(i + 1) * en));
    dNeg[i][n] = f2bf(__expf((float)(i + 1) * en));
  }
  if (t < 64) dQ[t] = __expf(-64.f * __expf(A_param[t]));
  __syncthreads();

  for (int ch = 0; ch < NCHUNK; ++ch) {
    // S0: build C~, B-hat, stage X (zero H on first chunk)
    for (int e = t; e < CHUNK * 64; e += 256) {
      int i = e >> 6, n = e & 63;
      long grow = (rowBase + ch * CHUNK + i) * XBC_W;
      float Bv = xBC[grow + 2 * D_INNER + n];
      float Cv = xBC[grow + 2 * D_INNER + D_STATE + n];
      Ach[i][n]  = f2bf(Cv * bf2f(dPos[i][n]));
      Btmp[i][n] = f2bf(Bv * bf2f(dNeg[i][n]));
    }
    for (int e = t; e < CHUNK * CG; e += 256) {
      int cc = e & (CG - 1);
      int j  = e >> 5;
      Xh[cc][64 + j] = f2bf(xconv[(rowBase + ch * CHUNK + j) * D_INNER + cg0 + cc]);
      if (ch == 0) Xh[cc][j] = 0;
    }
    __syncthreads();

    // S1: G = C~ @ Bhat^T (64x64x64), masked j<=i, into Ach[i][64+j]
#pragma unroll
    for (int s = 0; s < 2; ++s) {
      int tt = wave * 2 + s;
      int ti = tt >> 2, tj = tt & 3;
      v8f g = {};
#pragma unroll
      for (int kk = 0; kk < 2; ++kk) {
        V16 af, bfv;
        int mrow = ti * 16 + ql;
        af.q[0] = *(const u32x4*)&Ach[mrow][kk * 32 + hf * 16];
        af.q[1] = *(const u32x4*)&Ach[mrow][kk * 32 + hf * 16 + 8];
        int jrow = tj * 16 + ql;
        const u32x4* bp = (const u32x4*)&Btmp[jrow][kk * 32 + hf * 16];
        bfv.q[0] = bp[0]; bfv.q[1] = bp[1];
        g = __builtin_amdgcn_wmma_f32_16x16x32_bf16(false, af.v, false, bfv.v,
                                                    (short)0, g, false, false);
      }
#pragma unroll
      for (int r = 0; r < 8; ++r) {
        int gi = ti * 16 + r + 8 * hf;
        int gj = tj * 16 + ql;
        Ach[gi][64 + gj] = (gi >= gj) ? f2bf(g[r]) : (unsigned short)0;
      }
    }
    __syncthreads();

    // S2a: rebuild Btmp as Bbar^T[n][j] = B[j,n] * a_n^{63-j}
    for (int e = t; e < CHUNK * 64; e += 256) {
      int j = e >> 6, n = e & 63;
      long grow = (rowBase + ch * CHUNK + j) * XBC_W;
      float Bv = xBC[grow + 2 * D_INNER + n];
      Btmp[n][j] = f2bf(Bv * dQ[n] * bf2f(dNeg[j][n]));
    }
    // S2b: Y = [C~|G] @ [H;X]  (64 x CG, K=128), one 16x16 tile per wave
    {
      int ti = wave >> 1, tn = wave & 1;
      v8f y = {};
#pragma unroll
      for (int kk = 0; kk < 4; ++kk) {
        V16 af, bfv;
        int mrow = ti * 16 + ql;
        af.q[0] = *(const u32x4*)&Ach[mrow][kk * 32 + hf * 16];
        af.q[1] = *(const u32x4*)&Ach[mrow][kk * 32 + hf * 16 + 8];
        int crow = tn * 16 + ql;
        const u32x4* bp = (const u32x4*)&Xh[crow][kk * 32 + hf * 16];
        bfv.q[0] = bp[0]; bfv.q[1] = bp[1];
        y = __builtin_amdgcn_wmma_f32_16x16x32_bf16(false, af.v, false, bfv.v,
                                                    (short)0, y, false, false);
      }
#pragma unroll
      for (int r = 0; r < 8; ++r) {
        int gi = ti * 16 + r + 8 * hf;
        int gc = tn * 16 + ql;
        yssd[(rowBase + ch * CHUNK + gi) * D_INNER + cg0 + gc] = y[r];
      }
    }
    __syncthreads();

    // S3: H' = a^Q .* H + Bbar^T @ X  (64 x CG, K=64), RMW Xh[:, 0..63]
    {
      int ti = wave >> 1, tn = wave & 1;
      v8f hacc = {};
#pragma unroll
      for (int kk = 0; kk < 2; ++kk) {
        V16 af, bfv;
        int nrow = ti * 16 + ql;
        af.q[0] = *(const u32x4*)&Btmp[nrow][kk * 32 + hf * 16];
        af.q[1] = *(const u32x4*)&Btmp[nrow][kk * 32 + hf * 16 + 8];
        int crow = tn * 16 + ql;
        const u32x4* bp = (const u32x4*)&Xh[crow][64 + kk * 32 + hf * 16];
        bfv.q[0] = bp[0]; bfv.q[1] = bp[1];
        hacc = __builtin_amdgcn_wmma_f32_16x16x32_bf16(false, af.v, false, bfv.v,
                                                       (short)0, hacc, false, false);
      }
#pragma unroll
      for (int r = 0; r < 8; ++r) {
        int gn = ti * 16 + r + 8 * hf;
        int gc = tn * 16 + ql;
        float hold = bf2f(Xh[gc][gn]);
        Xh[gc][gn] = f2bf(hold * dQ[gn] + hacc[r]);
      }
    }
    __syncthreads();
  }
}

// ---------------------------------------------------------------------------
// yg = yssd * silu(z); LayerNorm over D_INNER; emit bf16 yn. One block/row.
// ---------------------------------------------------------------------------
__global__ __launch_bounds__(256)
void ln_gate(const float* __restrict__ yssd, const float* __restrict__ xBC,
             const float* __restrict__ lnw, const float* __restrict__ lnb,
             unsigned short* __restrict__ yn) {
  __shared__ float red[16];
  const long row = blockIdx.x;
  const int  t   = threadIdx.x;
  float v[4];
  float s1 = 0.f, s2 = 0.f;
#pragma unroll
  for (int k = 0; k < 4; ++k) {
    int c   = t + k * 256;
    float z = xBC[row * XBC_W + D_INNER + c];
    float g = yssd[row * D_INNER + c] * (z / (1.f + __expf(-z)));
    v[k] = g;
    s1 += g;
    s2 += g * g;
  }
#pragma unroll
  for (int off = 16; off; off >>= 1) {
    s1 += __shfl_xor(s1, off, 32);
    s2 += __shfl_xor(s2, off, 32);
  }
  if ((t & 31) == 0) { red[t >> 5] = s1; red[8 + (t >> 5)] = s2; }
  __syncthreads();
  float ts1 = 0.f, ts2 = 0.f;
#pragma unroll
  for (int w = 0; w < 8; ++w) { ts1 += red[w]; ts2 += red[8 + w]; }
  float mu   = ts1 * (1.f / 1024.f);
  float var  = ts2 * (1.f / 1024.f) - mu * mu;
  float rstd = rsqrtf(var + 1e-5f);
#pragma unroll
  for (int k = 0; k < 4; ++k) {
    int c = t + k * 256;
    yn[row * D_INNER + c] = f2bf((v[k] - mu) * rstd * lnw[c] + lnb[c]);
  }
}

// ---------------------------------------------------------------------------
extern "C" void kernel_launch(void* const* d_in, const int* in_sizes, int n_in,
                              void* d_out, int out_size, void* d_ws, size_t ws_size,
                              hipStream_t stream) {
  const float* x     = (const float*)d_in[0];
  const float* inw   = (const float*)d_in[1];
  const float* convw = (const float*)d_in[2];
  const float* convb = (const float*)d_in[3];
  const float* Ap    = (const float*)d_in[4];
  const float* lnw   = (const float*)d_in[5];
  const float* lnb   = (const float*)d_in[6];
  const float* outw  = (const float*)d_in[7];
  float* out = (float*)d_out;

  float* xBC   = (float*)d_ws;                          // 16384 x 2176 f32
  float* xconv = xBC + (size_t)NROW * XBC_W;            // 16384 x 1024 f32
  float* yssd  = xconv + (size_t)NROW * D_INNER;        // 16384 x 1024 f32
  unsigned short* xbf  = (unsigned short*)(yssd + (size_t)NROW * D_INNER);
  unsigned short* wbfi = xbf + (size_t)NROW * D_MODEL;  // 2176 x 1024 bf16
  unsigned short* wbfo = wbfi + (size_t)XBC_W * D_MODEL;// 1024 x 1024 bf16
  unsigned short* ynbf = xbf;                           // reuse after gemm1

  dim3 blk(256);
  // 0) one-time bf16 conversions (x, weights)
  cvt_bf16<<<dim3((NROW * D_MODEL) / 1024), blk, 0, stream>>>(x, xbf, (long)NROW * D_MODEL);
  cvt_bf16<<<dim3((XBC_W * D_MODEL) / 1024), blk, 0, stream>>>(inw, wbfi, (long)XBC_W * D_MODEL);
  cvt_bf16<<<dim3((D_MODEL * D_INNER) / 1024), blk, 0, stream>>>(outw, wbfo, (long)D_MODEL * D_INNER);
  // 1) xBC = x @ in_proj_w^T
  wmma_gemm_bf16<<<dim3(XBC_W / 128, NROW / 128), blk, 0, stream>>>(
      xbf, D_MODEL, wbfi, D_MODEL, xBC, XBC_W, D_MODEL);
  // 2) depthwise conv + SiLU
  conv_silu<<<dim3((NROW * D_INNER) / 256), blk, 0, stream>>>(xBC, convw, convb, xconv);
  // 3) chunked SSD scan (WMMA)
  ssd_scan<<<dim3(BATCH * NCG), blk, 0, stream>>>(xBC, xconv, Ap, yssd);
  // 4) gate + LayerNorm -> bf16 yn
  ln_gate<<<dim3(NROW), blk, 0, stream>>>(yssd, xBC, lnw, lnb, ynbf);
  // 5) out = yn @ out_proj_w^T
  wmma_gemm_bf16<<<dim3(D_MODEL / 128, NROW / 128), blk, 0, stream>>>(
      ynbf, D_INNER, wbfo, D_INNER, out, D_MODEL, D_INNER);
}
